// POS_predictorV2Chars_43602507989868
// MI455X (gfx1250) — matrix operand order repaired and checked by reading, not compiled, not run on hardware
//
#include <hip/hip_runtime.h>
#include <hip/hip_bf16.h>
#include <stdint.h>

// ---- problem dims (match reference) ----
#define B_      128
#define S_      256
#define L_      16
#define EMB_    256
#define HID_    512
#define NCLS_   20
#define NCHARS_ 128
#define CEMB_   64
#define CHID_   128
#define D_      (EMB_ + CHID_)   // 384 word-GRU input dim
#define G3C_    (3 * CHID_)      // 384 char gates
#define G3W_    (3 * HID_)       // 1536 word gates
#define KCAT_   (D_ + HID_)      // 896 fused K for word GRU

typedef __attribute__((ext_vector_type(16))) __bf16 v16bf;
typedef __attribute__((ext_vector_type(8)))  float  v8f;

union Frag16 { v16bf v; uint32_t u[8]; };

#define V8F_ZERO {0.f, 0.f, 0.f, 0.f, 0.f, 0.f, 0.f, 0.f}

__device__ __forceinline__ unsigned short f2bf_u(float f) {
  union { float f; uint32_t u; } a; a.f = f;
  uint32_t r = a.u + 0x7FFFu + ((a.u >> 16) & 1u);   // round-to-nearest-even
  return (unsigned short)(r >> 16);
}

__device__ __forceinline__ float sigmoidf_(float x) {
  return 1.0f / (1.0f + __expf(-x));
}

// Async DMA: 16 bytes global -> LDS per lane (ASYNCcnt-tracked).
// lds_off is a byte offset relative to the workgroup dynamic-LDS base.
__device__ __forceinline__ void async_load16(uint32_t lds_off, const void* gaddr) {
  asm volatile("global_load_async_to_lds_b128 %0, %1, off"
               :: "v"(lds_off), "v"(gaddr) : "memory");
}
#define WAIT_ASYNC(n) asm volatile("s_wait_asynccnt " #n ::: "memory")

// A tile 16x32 (MxK) bf16, row-major, row stride ld (elements). Two b128 loads.
// Lane L<16: M=L, K=0..15; lane L>=16: M=L-16, K=16..31.
__device__ __forceinline__ v16bf load_frag_a(const unsigned short* base, int ld) {
  const int lane = threadIdx.x & 31;
  const int row  = lane & 15;
  const int k    = (lane >> 4) << 4;          // 0 or 16
  const uint4* q = (const uint4*)(base + row * ld + k);
  uint4 lo = q[0];                            // K k..k+7
  uint4 hi = q[1];                            // K k+8..k+15
  Frag16 f;
  f.u[0] = lo.x; f.u[1] = lo.y; f.u[2] = lo.z; f.u[3] = lo.w;
  f.u[4] = hi.x; f.u[5] = hi.y; f.u[6] = hi.z; f.u[7] = hi.w;
  return f.v;
}

// B tile 32x16 (KxN) = W^T, W row-major [N][K]. Two b128 loads (32 contiguous B).
// Lane L<16: N=L, K=0..15; lane L>=16: N=L-16, K=16..31.
__device__ __forceinline__ v16bf load_frag_b(const unsigned short* base, int ld) {
  const int lane = threadIdx.x & 31;
  const int n    = lane & 15;
  const int k    = (lane >> 4) << 4;          // 0 or 16
  const uint4* q = (const uint4*)(base + n * ld + k);
  uint4 lo = q[0];
  uint4 hi = q[1];
  Frag16 f;
  f.u[0] = lo.x; f.u[1] = lo.y; f.u[2] = lo.z; f.u[3] = lo.w;
  f.u[4] = hi.x; f.u[5] = hi.y; f.u[6] = hi.z; f.u[7] = hi.w;
  return f.v;
}

// C/D 16x16 f32: VGPR r holds M = r + 8*(lane>=16), N = lane&15.
__device__ __forceinline__ void store_frag_c(float* base, int ld, v8f c) {
  const int lane = threadIdx.x & 31;
  const int n    = lane & 15;
  const int mo   = (lane >> 4) << 3;
#pragma unroll
  for (int r = 0; r < 8; ++r) base[(mo + r) * ld + n] = c[r];
}

__device__ __forceinline__ void store_frag_c_bias(float* base, int ld,
                                                  const float* bias_n0, v8f c) {
  const int lane = threadIdx.x & 31;
  const int n    = lane & 15;
  const int mo   = (lane >> 4) << 3;
  const float bv = bias_n0[n];
#pragma unroll
  for (int r = 0; r < 8; ++r) base[(mo + r) * ld + n] = c[r] + bv;
}

// ------------------------------------------------------------------
// Prep: proj[c,g] = char_emb_w[c] . char_Wi[g] + char_bi[g]  (fp32, 128x384)
// ------------------------------------------------------------------
__global__ void __launch_bounds__(256) k_prep_proj(
    const float* __restrict__ cew, const float* __restrict__ cWi,
    const float* __restrict__ cbi, float* __restrict__ proj) {
  int idx = blockIdx.x * blockDim.x + threadIdx.x;
  if (idx >= NCHARS_ * G3C_) return;
  int c = idx / G3C_, g = idx - c * G3C_;
  const float* e = cew + c * CEMB_;
  const float* w = cWi + g * CEMB_;
  float s = cbi[g];
#pragma unroll 8
  for (int d = 0; d < CEMB_; ++d) s += e[d] * w[d];
  proj[idx] = s;
}

// ------------------------------------------------------------------
// Prep: Wcat bf16 [1536][896] = [gru_Wi | gru_Wh], clsW padded bf16 [32][512]
// ------------------------------------------------------------------
__global__ void __launch_bounds__(256) k_prep_weights(
    const float* __restrict__ gWi, const float* __restrict__ gWh,
    const float* __restrict__ clsW,
    unsigned short* __restrict__ Wcat, unsigned short* __restrict__ clsWp) {
  const int total1 = G3W_ * KCAT_;
  const int total  = total1 + 32 * HID_;
  const int stride = gridDim.x * blockDim.x;
  for (int idx = blockIdx.x * blockDim.x + threadIdx.x; idx < total; idx += stride) {
    if (idx < total1) {
      int g = idx / KCAT_, k = idx - g * KCAT_;
      float v = (k < D_) ? gWi[g * D_ + k] : gWh[g * HID_ + (k - D_)];
      Wcat[idx] = f2bf_u(v);
    } else {
      int j = idx - total1;
      int n = j / HID_, k = j - n * HID_;
      float v = (n < NCLS_) ? clsW[n * HID_ + k] : 0.0f;
      clsWp[j] = f2bf_u(v);
    }
  }
}

// ------------------------------------------------------------------
// feats[:, 0:256] = bf16(emb_w[x])  (row r = b*S + s), float4 vectorized
// ------------------------------------------------------------------
__global__ void __launch_bounds__(256) k_feats_emb(
    const int* __restrict__ x, const float* __restrict__ embw,
    unsigned short* __restrict__ feats) {
  const int total  = B_ * S_ * (EMB_ / 4);
  const int stride = gridDim.x * blockDim.x;
  for (int idx = blockIdx.x * blockDim.x + threadIdx.x; idx < total; idx += stride) {
    int r = idx >> 6, e4 = idx & 63;       // EMB_/4 = 64
    int tok = x[r];
    float4 v = ((const float4*)(embw + tok * EMB_))[e4];
    uint2 o;
    o.x = (uint32_t)f2bf_u(v.x) | ((uint32_t)f2bf_u(v.y) << 16);
    o.y = (uint32_t)f2bf_u(v.z) | ((uint32_t)f2bf_u(v.w) << 16);
    *(uint2*)(feats + r * D_ + e4 * 4) = o;
  }
}

// ------------------------------------------------------------------
// Char GRU: 16 sequences per WG, 16 timesteps.
// Wh tiles live in REGISTERS (12 bf16 fragments/wave, loop-invariant).
// xw rows (proj gather) double-buffered in LDS via async DMA.
// ------------------------------------------------------------------
// LDS layout (bytes): xw0 @0 (24576) | xw1 @24576 (24576) | gh @49152 (24576)
//                     hb @73728 (4096) | hf @77824 (8192)   => total 86016
#define CG_XW0 0u
#define CG_XW1 24576u
#define CG_GH  49152u
#define CG_HB  73728u
#define CG_HF  77824u

__global__ void __launch_bounds__(256) k_char_gru(
    const int* __restrict__ xch, const float* __restrict__ cWh,
    const float* __restrict__ cbh, const float* __restrict__ proj,
    unsigned short* __restrict__ feats) {
  extern __shared__ unsigned char smem[];
  float*          gh_s = (float*)(smem + CG_GH);
  unsigned short* hb_s = (unsigned short*)(smem + CG_HB);
  float*          hf_s = (float*)(smem + CG_HF);

  const int tid  = threadIdx.x;
  const int wave = tid >> 5;
  const int lane = tid & 31;
  const int n0   = blockIdx.x * 16;   // flat word index n = s*B + b

  // Hoist B fragments: 3 N-tiles x 4 K-steps of Wh^T, fp32 -> bf16, registers.
  Frag16 Bf[12];
  {
    const int n  = lane & 15;
    const int kh = (lane >> 4) << 4;  // 0 or 16
#pragma unroll
    for (int i = 0; i < 3; ++i) {
      const float* wrow = cWh + ((wave * 3 + i) * 16 + n) * CHID_ + kh;
#pragma unroll
      for (int kk = 0; kk < 4; ++kk) {
        const float* src = wrow + kk * 32;
#pragma unroll
        for (int j = 0; j < 8; ++j) {
          Bf[i * 4 + kk].u[j] = (uint32_t)f2bf_u(src[2 * j]) |
                                ((uint32_t)f2bf_u(src[2 * j + 1]) << 16);
        }
      }
    }
  }

  for (int i = tid; i < 16 * CHID_; i += 256) { hb_s[i] = 0; hf_s[i] = 0.0f; }

  // stage xw(t=0): 16 rows x 96 chunks of 16B from proj[char] rows (6 asyncs/lane)
  for (int c = tid; c < 16 * 96; c += 256) {
    int m = c / 96, q = c - m * 96;
    int ch = xch[(n0 + m) * L_ + 0];
    async_load16(CG_XW0 + (uint32_t)(m * (G3C_ * 4) + q * 16),
                 proj + ch * G3C_ + q * 4);
  }
  __syncthreads();

  for (int t = 0; t < L_; ++t) {
    const uint32_t curOff = (t & 1) ? CG_XW1 : CG_XW0;
    const float*   xw_s   = (const float*)(smem + curOff);
    // prefetch xw(t+1) into the other buffer (overlaps GEMM + gates)
    if (t + 1 < L_) {
      const uint32_t nxtOff = (t & 1) ? CG_XW0 : CG_XW1;
      for (int c = tid; c < 16 * 96; c += 256) {
        int m = c / 96, q = c - m * 96;
        int ch = xch[(n0 + m) * L_ + (t + 1)];
        async_load16(nxtOff + (uint32_t)(m * (G3C_ * 4) + q * 16),
                     proj + ch * G3C_ + q * 4);
      }
    }
    // gh = h @ Wh^T : k-outer, 1 A-load feeds 3 WMMAs
    v8f acc[3] = {V8F_ZERO, V8F_ZERO, V8F_ZERO};
#pragma unroll
    for (int kk = 0; kk < 4; ++kk) {
      v16bf a = load_frag_a(hb_s + kk * 32, CHID_);
#pragma unroll
      for (int i = 0; i < 3; ++i)
        acc[i] = __builtin_amdgcn_wmma_f32_16x16x32_bf16(false, a, false,
                     Bf[i * 4 + kk].v, (short)0, acc[i], false, false);
    }
#pragma unroll
    for (int i = 0; i < 3; ++i)
      store_frag_c(gh_s + (wave * 3 + i) * 16, G3C_, acc[i]);
    // drain xw(t) (in-order completion: <=6 outstanding leaves only t+1's)
    if (t + 1 < L_) { WAIT_ASYNC(0x6); } else { WAIT_ASYNC(0x0); }
    __syncthreads();
    // gates
    for (int i = tid; i < 16 * CHID_; i += 256) {
      int m = i >> 7, j = i & (CHID_ - 1);
      float ghr = gh_s[m * G3C_ + j]             + cbh[j];
      float ghz = gh_s[m * G3C_ + CHID_ + j]     + cbh[CHID_ + j];
      float ghn = gh_s[m * G3C_ + 2 * CHID_ + j] + cbh[2 * CHID_ + j];
      float r  = sigmoidf_(xw_s[m * G3C_ + j] + ghr);
      float z  = sigmoidf_(xw_s[m * G3C_ + CHID_ + j] + ghz);
      float nn = tanhf(xw_s[m * G3C_ + 2 * CHID_ + j] + r * ghn);
      float hn = (1.0f - z) * nn + z * hf_s[i];
      hf_s[i] = hn;
      hb_s[i] = f2bf_u(hn);
    }
    __syncthreads();
  }
  // h_T -> feats[:, 256:384]; n = s*B + b  =>  b = n&127, s = n>>7
  for (int i = tid; i < 16 * CHID_; i += 256) {
    int m = i >> 7, j = i & (CHID_ - 1);
    int n = n0 + m;
    int b = n & (B_ - 1), s = n >> 7;
    feats[(b * S_ + s) * D_ + EMB_ + j] = hb_s[i];
  }
}

// ------------------------------------------------------------------
// Word GRU: 16 batch rows per WG, 256 timesteps.
// Fused [feats_t | h] (16x896) @ Wcat^T, split accumulation at k=384.
// feats_t tile double-buffered in LDS via async DMA; k-outer GEMM
// (1 A-load -> 12 WMMAs), Wcat streamed from L2 with global_load_b128.
// ------------------------------------------------------------------
// LDS (bytes): A0 @0 (28672) | A1 @28672 (28672) | gx @57344 (98304)
//   gh @155648 (98304) | bi @253952 (6144) | bh @260096 (6144)
//   hf @266240 (32768)  => total 299008
#define WG_A0 0u
#define WG_A1 28672u
#define WG_GX 57344u
#define WG_GH 155648u
#define WG_BI 253952u
#define WG_BH 260096u
#define WG_HF 266240u

__global__ void __launch_bounds__(256) k_word_gru(
    const unsigned short* __restrict__ feats,
    const unsigned short* __restrict__ Wcat,
    const float* __restrict__ gbi, const float* __restrict__ gbh,
    unsigned short* __restrict__ gru_out) {
  extern __shared__ unsigned char smem[];
  unsigned short* A0_s = (unsigned short*)(smem + WG_A0);
  unsigned short* A1_s = (unsigned short*)(smem + WG_A1);
  float*          gx_s = (float*)(smem + WG_GX);
  float*          gh_s = (float*)(smem + WG_GH);
  float*          bi_s = (float*)(smem + WG_BI);
  float*          bh_s = (float*)(smem + WG_BH);
  float*          hf_s = (float*)(smem + WG_HF);

  const int tid  = threadIdx.x;
  const int wave = tid >> 5;
  const int b0   = blockIdx.x * 16;

  for (int i = tid; i < G3W_; i += 256) { bi_s[i] = gbi[i]; bh_s[i] = gbh[i]; }
  for (int i = tid; i < 16 * HID_; i += 256) {
    hf_s[i] = 0.0f;
    int m = i >> 9, j = i & (HID_ - 1);
    A0_s[m * KCAT_ + D_ + j] = 0;
    A1_s[m * KCAT_ + D_ + j] = 0;
  }
  // stage feats(s=0) into A0: 16 rows x 48 chunks of 16B (3 asyncs/lane)
  for (int c = tid; c < 16 * 48; c += 256) {
    int m = c / 48, q = c - m * 48;
    async_load16(WG_A0 + (uint32_t)(m * (KCAT_ * 2) + q * 16),
                 feats + ((size_t)(b0 + m) * S_ + 0) * D_ + q * 8);
  }
  __syncthreads();

  for (int s = 0; s < S_; ++s) {
    const unsigned short* Acur = (s & 1) ? A1_s : A0_s;
    // prefetch feats(s+1) into the other buffer
    if (s + 1 < S_) {
      const uint32_t nxtOff = (s & 1) ? WG_A0 : WG_A1;
      for (int c = tid; c < 16 * 48; c += 256) {
        int m = c / 48, q = c - m * 48;
        async_load16(nxtOff + (uint32_t)(m * (KCAT_ * 2) + q * 16),
                     feats + ((size_t)(b0 + m) * S_ + (s + 1)) * D_ + q * 8);
      }
      WAIT_ASYNC(0x3);   // drain feats(s); leaves only (s+1)'s in flight
    } else {
      WAIT_ASYNC(0x0);
    }
    __syncthreads();

    // 96 N-tiles over 8 waves, 12 accumulators, k-outer
    v8f acc[12];
#pragma unroll
    for (int i = 0; i < 12; ++i) acc[i] = (v8f)V8F_ZERO;
    for (int k0 = 0; k0 < D_; k0 += 32) {
      v16bf a = load_frag_a(Acur + k0, KCAT_);
#pragma unroll
      for (int i = 0; i < 12; ++i) {
        v16bf b = load_frag_b(Wcat + (size_t)(wave * 12 + i) * 16 * KCAT_ + k0, KCAT_);
        acc[i] = __builtin_amdgcn_wmma_f32_16x16x32_bf16(false, a, false, b,
                                                         (short)0, acc[i], false, false);
      }
    }
#pragma unroll
    for (int i = 0; i < 12; ++i) {
      int nt = wave * 12 + i;
      store_frag_c_bias(gx_s + nt * 16, G3W_, bi_s + nt * 16, acc[i]);
      acc[i] = (v8f)V8F_ZERO;
    }
    for (int k0 = D_; k0 < KCAT_; k0 += 32) {
      v16bf a = load_frag_a(Acur + k0, KCAT_);
#pragma unroll
      for (int i = 0; i < 12; ++i) {
        v16bf b = load_frag_b(Wcat + (size_t)(wave * 12 + i) * 16 * KCAT_ + k0, KCAT_);
        acc[i] = __builtin_amdgcn_wmma_f32_16x16x32_bf16(false, a, false, b,
                                                         (short)0, acc[i], false, false);
      }
    }
#pragma unroll
    for (int i = 0; i < 12; ++i) {
      int nt = wave * 12 + i;
      store_frag_c_bias(gh_s + nt * 16, G3W_, bh_s + nt * 16, acc[i]);
    }
    __syncthreads();

    // gates + h update; h bf16 into BOTH A buffers and gru_out
    for (int i = tid; i < 16 * HID_; i += 256) {
      int m = i >> 9, j = i & (HID_ - 1);
      float r  = sigmoidf_(gx_s[m * G3W_ + j] + gh_s[m * G3W_ + j]);
      float z  = sigmoidf_(gx_s[m * G3W_ + HID_ + j] + gh_s[m * G3W_ + HID_ + j]);
      float nn = tanhf(gx_s[m * G3W_ + 2 * HID_ + j] + r * gh_s[m * G3W_ + 2 * HID_ + j]);
      float hn = (1.0f - z) * nn + z * hf_s[i];
      hf_s[i] = hn;
      unsigned short hb = f2bf_u(hn);
      A0_s[m * KCAT_ + D_ + j] = hb;
      A1_s[m * KCAT_ + D_ + j] = hb;
      gru_out[((size_t)(b0 + m) * S_ + s) * HID_ + j] = hb;
    }
    __syncthreads();
  }
}

// ------------------------------------------------------------------
// Classifier: out[r, 0:20] = gru_out[r] @ clsW^T + clsb via wmma (N padded to 32)
// ------------------------------------------------------------------
__global__ void __launch_bounds__(32) k_classifier(
    const unsigned short* __restrict__ gru_out,
    const unsigned short* __restrict__ clsWp,
    const float* __restrict__ clsb, float* __restrict__ out) {
  const int m0 = blockIdx.x * 16;
  v8f acc0 = V8F_ZERO;
  v8f acc1 = V8F_ZERO;
#pragma unroll
  for (int k0 = 0; k0 < HID_; k0 += 32) {
    v16bf a  = load_frag_a(gru_out + (size_t)m0 * HID_ + k0, HID_);
    v16bf b0 = load_frag_b(clsWp + k0, HID_);
    v16bf b1 = load_frag_b(clsWp + 16 * HID_ + k0, HID_);
    acc0 = __builtin_amdgcn_wmma_f32_16x16x32_bf16(false, a, false, b0,
                                                   (short)0, acc0, false, false);
    acc1 = __builtin_amdgcn_wmma_f32_16x16x32_bf16(false, a, false, b1,
                                                   (short)0, acc1, false, false);
  }
  const int lane = threadIdx.x & 31;
  const int n    = lane & 15;
  const int mo   = (lane >> 4) << 3;
#pragma unroll
  for (int r = 0; r < 8; ++r)
    out[(m0 + mo + r) * NCLS_ + n] = acc0[r] + clsb[n];
  if (n + 16 < NCLS_) {
#pragma unroll
    for (int r = 0; r < 8; ++r)
      out[(m0 + mo + r) * NCLS_ + n + 16] = acc1[r] + clsb[n + 16];
  }
}

// ------------------------------------------------------------------
extern "C" void kernel_launch(void* const* d_in, const int* in_sizes, int n_in,
                              void* d_out, int out_size, void* d_ws, size_t ws_size,
                              hipStream_t stream) {
  (void)in_sizes; (void)n_in; (void)out_size; (void)ws_size;
  const int*   x       = (const int*)  d_in[0];
  const int*   x_chars = (const int*)  d_in[1];
  const float* emb_w   = (const float*)d_in[2];
  const float* cew     = (const float*)d_in[3];
  const float* cWi     = (const float*)d_in[4];
  const float* cWh     = (const float*)d_in[5];
  const float* cbi     = (const float*)d_in[6];
  const float* cbh     = (const float*)d_in[7];
  const float* gWi     = (const float*)d_in[8];
  const float* gWh     = (const float*)d_in[9];
  const float* gbi     = (const float*)d_in[10];
  const float* gbh     = (const float*)d_in[11];
  const float* clsW    = (const float*)d_in[12];
  const float* clsb    = (const float*)d_in[13];
  float* out = (float*)d_out;

  // workspace layout (bytes)
  unsigned char* ws = (unsigned char*)d_ws;
  float*          proj  = (float*)(ws + 0);                  // 128*384*4   = 196608
  unsigned short* Wcat  = (unsigned short*)(ws + 196608);    // 1536*896*2  = 2752512
  unsigned short* clsWp = (unsigned short*)(ws + 2949120);   // 32*512*2    = 32768
  unsigned short* feats = (unsigned short*)(ws + 2981888);   // 32768*384*2 = 25165824
  unsigned short* gout  = (unsigned short*)(ws + 28147712);  // 32768*512*2 = 33554432
                                                             // total ~61.7 MB

  k_prep_proj   <<<(NCHARS_ * G3C_ + 255) / 256, 256, 0, stream>>>(cew, cWi, cbi, proj);
  k_prep_weights<<<1024, 256, 0, stream>>>(gWi, gWh, clsW, Wcat, clsWp);
  k_feats_emb   <<<2048, 256, 0, stream>>>(x, emb_w, feats);
  k_char_gru    <<<(B_ * S_) / 16, 256, 86016, stream>>>(x_chars, cWh, cbh, proj, feats);
  k_word_gru    <<<B_ / 16, 256, 299008, stream>>>(feats, Wcat, gbi, gbh, gout);
  k_classifier  <<<(B_ * S_) / 16, 32, 0, stream>>>(gout, clsWp, clsb, out);
}
